// SoftBERTScoreLoss_58411555226044
// MI455X (gfx1250) — compile-verified, matrix operand order checked by reading, not run on hardware
//
#include <hip/hip_runtime.h>

#define IGNORE_INDEX (-100)
#define NEG_INF (-1e9f)

static constexpr int NB = 8;        // batch
static constexpr int NT = 512;      // tokens
static constexpr int NV = 32128;    // vocab
static constexpr int ND = 512;      // embed dim
static constexpr int NM = NB * NT;  // 4096 rows

typedef __attribute__((ext_vector_type(16))) __bf16 v16bf;
typedef __attribute__((ext_vector_type(8)))  __bf16 v8bf;
typedef __attribute__((ext_vector_type(2)))  __bf16 v2bf;
typedef __attribute__((ext_vector_type(8)))  float  v8f;

__device__ __forceinline__ v16bf load_frag16(const __bf16* lo, const __bf16* hi) {
    v8bf a0 = *(const v8bf*)lo;
    v8bf a1 = *(const v8bf*)hi;
    v16bf r;
#pragma unroll
    for (int i = 0; i < 8; ++i) { r[i] = a0[i]; r[i + 8] = a1[i]; }
    return r;
}

__device__ __forceinline__ void atomicMaxF(float* addr, float val) {
    unsigned int* a = (unsigned int*)addr;
    unsigned int old = __float_as_uint(*addr);
    while (__uint_as_float(old) < val) {
        unsigned int assumed = old;
        old = atomicCAS(a, assumed, __float_as_uint(val));
        if (old == assumed) break;
    }
}

// ---------------------------------------------------------------- init
__global__ void init_kernel(float* simcolmax) {
    int i = blockIdx.x * blockDim.x + threadIdx.x;
    if (i < NM) simcolmax[i] = NEG_INF;
}

// ------------------------------------------------- softmax row stats
__global__ __launch_bounds__(256) void row_stats_kernel(
    const float* __restrict__ logits, float* __restrict__ rowmax,
    float* __restrict__ rowsum) {
    const int row = blockIdx.x;
    const int tid = threadIdx.x;
    const float* x = logits + (size_t)row * NV;
    float m = -3.0e38f, s = 0.0f;
    for (int v = tid * 4; v < NV; v += 256 * 4) {
        float4 f = *(const float4*)(x + v);
        float mx = fmaxf(fmaxf(f.x, f.y), fmaxf(f.z, f.w));
        if (mx > m) { s *= __expf(m - mx); m = mx; }
        s += __expf(f.x - m) + __expf(f.y - m) + __expf(f.z - m) + __expf(f.w - m);
    }
    __shared__ float sm[256], ss[256];
    sm[tid] = m; ss[tid] = s;
    __syncthreads();
    for (int st = 128; st > 0; st >>= 1) {
        if (tid < st) {
            float m2 = sm[tid + st], s2 = ss[tid + st];
            float mm = fmaxf(sm[tid], m2);
            ss[tid] = ss[tid] * __expf(sm[tid] - mm) + s2 * __expf(m2 - mm);
            sm[tid] = mm;
        }
        __syncthreads();
    }
    if (tid == 0) { rowmax[row] = sm[0]; rowsum[row] = ss[0]; }
}

// ---------------------- fused soft_pred = softmax(logits) @ E  (WMMA bf16)
// block tile 64(M) x 128(N); 8 waves: wave -> (mi = w&3) 16-row slab,
// (ng = w>>2) 64-col half, 4 accumulators. Double-buffered LDS so the
// global loads for tile k+1 overlap the WMMAs of tile k (one barrier/iter).
__global__ __launch_bounds__(256) void softpred_gemm_kernel(
    const float* __restrict__ logits, const float* __restrict__ emb,
    const float* __restrict__ rowmax, const float* __restrict__ rowsum,
    float* __restrict__ soft_pred) {
    __shared__ __bf16 As[2][64][32];    // [buf][m][k]   8 KB
    __shared__ __bf16 Bs[2][128][32];   // [buf][n][k]  16 KB (transposed)
    __shared__ float sMax[64], sInv[64];

    const int tid = threadIdx.x;
    const int m0 = blockIdx.x * 64;
    const int n0 = blockIdx.y * 128;
    if (tid < 64) {
        sMax[tid] = rowmax[m0 + tid];
        sInv[tid] = 1.0f / rowsum[m0 + tid];
    }
    __syncthreads();

    const int wave = tid >> 5, lane = tid & 31;
    const int mi = wave & 3, ng = wave >> 2;
    const int half = lane >> 4, l16 = lane & 15;

    // A staging: thread -> row ar, 8-wide k chunk
    const int ar = tid >> 2, ac = (tid & 3) * 8;
    const float rm = sMax[ar];
    // B staging: thread -> even k pair kb, two 4-wide d chunks
    const int kb = (tid >> 4) * 2;
    const int dgb = (tid & 15) * 4;

    auto stage = [&](int sb, int kk) {
        // ---- A: exp(logit - rowmax) -> bf16, one b128 LDS store
        {
            const float* src = logits + (size_t)(m0 + ar) * NV + kk + ac;
            float4 f0 = *(const float4*)(src);
            float4 f1 = *(const float4*)(src + 4);
            v8bf o;
            o[0] = (__bf16)__expf(f0.x - rm); o[1] = (__bf16)__expf(f0.y - rm);
            o[2] = (__bf16)__expf(f0.z - rm); o[3] = (__bf16)__expf(f0.w - rm);
            o[4] = (__bf16)__expf(f1.x - rm); o[5] = (__bf16)__expf(f1.y - rm);
            o[6] = (__bf16)__expf(f1.z - rm); o[7] = (__bf16)__expf(f1.w - rm);
            *(v8bf*)&As[sb][ar][ac] = o;
        }
        // ---- B: E[k][d],E[k+1][d] -> packed bf16x2 at Bs[d][k..k+1]
        {
            const float* s0 = emb + (size_t)(kk + kb) * ND + n0;
#pragma unroll
            for (int u = 0; u < 2; ++u) {
                const int d0 = dgb + u * 64;
                float4 f0 = *(const float4*)(s0 + d0);
                float4 f1 = *(const float4*)(s0 + ND + d0);
                v2bf p0; p0[0] = (__bf16)f0.x; p0[1] = (__bf16)f1.x;
                v2bf p1; p1[0] = (__bf16)f0.y; p1[1] = (__bf16)f1.y;
                v2bf p2; p2[0] = (__bf16)f0.z; p2[1] = (__bf16)f1.z;
                v2bf p3; p3[0] = (__bf16)f0.w; p3[1] = (__bf16)f1.w;
                *(v2bf*)&Bs[sb][d0 + 0][kb] = p0;
                *(v2bf*)&Bs[sb][d0 + 1][kb] = p1;
                *(v2bf*)&Bs[sb][d0 + 2][kb] = p2;
                *(v2bf*)&Bs[sb][d0 + 3][kb] = p3;
            }
        }
    };

    v8f acc[4] = {};
    int buf = 0;
    stage(0, 0);
    __syncthreads();

    const int arow = mi * 16 + l16;
    for (int kk = 0; kk < NV; kk += 32) {
        const int nbuf = buf ^ 1;
        if (kk + 32 < NV) stage(nbuf, kk + 32);   // overlaps WMMAs below

        v16bf a = load_frag16(&As[buf][arow][half * 8],
                              &As[buf][arow][16 + half * 8]);
#pragma unroll
        for (int nj = 0; nj < 4; ++nj) {
            const int col = ng * 64 + nj * 16 + l16;
            v16bf b = load_frag16(&Bs[buf][col][half * 8],
                                  &Bs[buf][col][16 + half * 8]);
            acc[nj] = __builtin_amdgcn_wmma_f32_16x16x32_bf16(
                false, a, false, b, (short)0, acc[nj], false, false);
        }
        __syncthreads();
        buf = nbuf;
    }

#pragma unroll
    for (int v = 0; v < 8; ++v) {
        const int rl = mi * 16 + v + 8 * half;
        const float inv = sInv[rl];
        const size_t row = (size_t)(m0 + rl);
#pragma unroll
        for (int nj = 0; nj < 4; ++nj) {
            const int col = n0 + ng * 64 + nj * 16 + l16;
            soft_pred[row * ND + col] = acc[nj][v] * inv;
        }
    }
}

// --------------------- L2-normalize soft_pred; gather+normalize E[labels]
__global__ __launch_bounds__(512) void norm_gather_kernel(
    const float* __restrict__ soft_pred, const float* __restrict__ emb,
    const int* __restrict__ labels, __bf16* __restrict__ p_n,
    __bf16* __restrict__ r_n, int* __restrict__ valid) {
    const int m = blockIdx.x;
    const int d = threadIdx.x;
    __shared__ float red[512];

    float x = soft_pred[(size_t)m * ND + d];
    red[d] = x * x;
    __syncthreads();
    for (int st = 256; st > 0; st >>= 1) {
        if (d < st) red[d] += red[d + st];
        __syncthreads();
    }
    if (d == 0) red[0] = 1.0f / fmaxf(sqrtf(red[0]), 1e-12f);
    __syncthreads();
    const float inv1 = red[0];
    p_n[(size_t)m * ND + d] = (__bf16)(x * inv1);

    const int lab = labels[m];
    const int ok = (lab != IGNORE_INDEX) ? 1 : 0;
    const int sl = ok ? lab : 0;
    const float r = emb[(size_t)sl * ND + d];
    __syncthreads();
    red[d] = r * r;
    __syncthreads();
    for (int st = 256; st > 0; st >>= 1) {
        if (d < st) red[d] += red[d + st];
        __syncthreads();
    }
    if (d == 0) red[0] = 1.0f / fmaxf(sqrtf(red[0]), 1e-12f);
    __syncthreads();
    r_n[(size_t)m * ND + d] = (__bf16)(r * red[0]);
    if (d == 0) valid[m] = ok;
}

// -------- sim = p_n @ r_n^T per batch (WMMA bf16), masked row/col maxes.
// grid (4 i-blocks of 128 rows, 8 batches); wave owns one 16-row i-tile,
// processes 4 j-tiles per group so each A fragment feeds 4 WMMAs.
__global__ __launch_bounds__(256) void sim_kernel(
    const __bf16* __restrict__ p_n, const __bf16* __restrict__ r_n,
    const int* __restrict__ valid, float* __restrict__ simrowmax,
    float* __restrict__ simcolmax) {
    const int b = blockIdx.y;
    const int wave = threadIdx.x >> 5, lane = threadIdx.x & 31;
    const int half = lane >> 4, l16 = lane & 15;
    const int i0 = (blockIdx.x * 8 + wave) * 16;
    const size_t base = (size_t)b * NT * ND;
    const int vbase = b * NT;

    int vi[8];
#pragma unroll
    for (int v = 0; v < 8; ++v) vi[v] = valid[vbase + i0 + v + 8 * half];

    float rmax[8];
#pragma unroll
    for (int v = 0; v < 8; ++v) rmax[v] = NEG_INF;

    const __bf16* arow = p_n + base + (size_t)(i0 + l16) * ND;

    for (int jg = 0; jg < 8; ++jg) {            // groups of 4 j-tiles
        v8f acc[4] = {};
        const __bf16* brow[4];
#pragma unroll
        for (int jj = 0; jj < 4; ++jj)
            brow[jj] = r_n + base + (size_t)((jg * 4 + jj) * 16 + l16) * ND;

        for (int k = 0; k < ND; k += 32) {
            v16bf a = load_frag16(arow + k + half * 8, arow + k + 16 + half * 8);
#pragma unroll
            for (int jj = 0; jj < 4; ++jj) {
                v16bf bb = load_frag16(brow[jj] + k + half * 8,
                                       brow[jj] + k + 16 + half * 8);
                acc[jj] = __builtin_amdgcn_wmma_f32_16x16x32_bf16(
                    false, a, false, bb, (short)0, acc[jj], false, false);
            }
        }
#pragma unroll
        for (int jj = 0; jj < 4; ++jj) {
            const int jrow = (jg * 4 + jj) * 16 + l16;
            const int vj = valid[vbase + jrow];
            float cmax = NEG_INF;
#pragma unroll
            for (int v = 0; v < 8; ++v) {
                const float c = acc[jj][v];
                rmax[v] = fmaxf(rmax[v], vj ? c : NEG_INF);  // mask invalid cols
                cmax = fmaxf(cmax, vi[v] ? c : NEG_INF);     // mask invalid rows
            }
            cmax = fmaxf(cmax, __shfl_xor(cmax, 16, 32));    // merge halves
            if (lane < 16) atomicMaxF(&simcolmax[vbase + jrow], cmax);
        }
    }
    // reduce row maxes across the 16 lanes that share each row (within a half)
#pragma unroll
    for (int v = 0; v < 8; ++v) {
        float r = rmax[v];
        r = fmaxf(r, __shfl_xor(r, 1, 32));
        r = fmaxf(r, __shfl_xor(r, 2, 32));
        r = fmaxf(r, __shfl_xor(r, 4, 32));
        r = fmaxf(r, __shfl_xor(r, 8, 32));
        if (l16 == 0) simrowmax[vbase + i0 + v + 8 * half] = r;
    }
}

// ---------------------------------------------------------- finalize
__global__ __launch_bounds__(512) void finalize_kernel(
    const int* __restrict__ valid, const float* __restrict__ simrowmax,
    const float* __restrict__ simcolmax, float* __restrict__ out) {
    __shared__ float sc[512], sp[512], sr[512];
    const int t = threadIdx.x;
    float loss = 0.0f;
    for (int b = 0; b < NB; ++b) {
        const int m = b * NT + t;
        const int ok = valid[m];
        sc[t] = (float)ok;
        sp[t] = ok ? simrowmax[m] : 0.0f;
        sr[t] = ok ? simcolmax[m] : 0.0f;
        __syncthreads();
        for (int st = 256; st > 0; st >>= 1) {
            if (t < st) { sc[t] += sc[t + st]; sp[t] += sp[t + st]; sr[t] += sr[t + st]; }
            __syncthreads();
        }
        if (t == 0) {
            const float cnt = sc[0];
            const float dc = fmaxf(cnt, 1.0f);
            const float prec = sp[0] / dc;
            const float rec = sr[0] / dc;
            const float den = prec + rec;
            const float f1 = (den > 0.0f) ? (2.0f * prec * rec / fmaxf(den, 1e-8f)) : 0.0f;
            loss += (cnt > 0.0f) ? (1.0f - f1) : 0.0f;
        }
        __syncthreads();
    }
    if (t == 0) out[0] = loss / (float)NB;
}

// ---------------------------------------------------------------- launch
extern "C" void kernel_launch(void* const* d_in, const int* in_sizes, int n_in,
                              void* d_out, int out_size, void* d_ws, size_t ws_size,
                              hipStream_t stream) {
    const float* logits = (const float*)d_in[0];
    const int*   labels = (const int*)d_in[1];
    const float* emb    = (const float*)d_in[2];
    float* out = (float*)d_out;

    char* ws = (char*)d_ws;
    size_t off = 0;
    float* rowmax    = (float*)(ws + off); off += (size_t)NM * 4;               // 16 KB
    float* rowsum    = (float*)(ws + off); off += (size_t)NM * 4;               // 16 KB
    float* soft_pred = (float*)(ws + off); off += (size_t)NM * ND * 4;          // 8 MB
    __bf16* p_n      = (__bf16*)(ws + off); off += (size_t)NM * ND * 2;         // 4 MB
    __bf16* r_n      = (__bf16*)(ws + off); off += (size_t)NM * ND * 2;         // 4 MB
    int*   valid     = (int*)(ws + off); off += (size_t)NM * 4;
    float* simrowmax = (float*)(ws + off); off += (size_t)NM * 4;
    float* simcolmax = (float*)(ws + off); off += (size_t)NM * 4;

    init_kernel<<<(NM + 255) / 256, 256, 0, stream>>>(simcolmax);
    row_stats_kernel<<<NM, 256, 0, stream>>>(logits, rowmax, rowsum);
    softpred_gemm_kernel<<<dim3(NM / 64, ND / 128), 256, 0, stream>>>(
        logits, emb, rowmax, rowsum, soft_pred);
    norm_gather_kernel<<<NM, 512, 0, stream>>>(soft_pred, emb, labels, p_n, r_n, valid);
    sim_kernel<<<dim3(NT / 128, NB), 256, 0, stream>>>(p_n, r_n, valid,
                                                       simrowmax, simcolmax);
    finalize_kernel<<<1, 512, 0, stream>>>(valid, simrowmax, simcolmax, out);
}
